// DomainAdaptiveContrastiveLoss_71665824301677
// MI455X (gfx1250) — matrix operand order repaired and checked by reading, not compiled, not run on hardware
//
#include <hip/hip_runtime.h>
#include <hip/hip_bf16.h>

typedef __attribute__((ext_vector_type(16))) _Float16 v16h;
typedef __attribute__((ext_vector_type(8)))  float    v8f;
typedef __attribute__((ext_vector_type(4)))  float    f32x4;
typedef __attribute__((ext_vector_type(4)))  _Float16 h16x4;

#define DCONST   128
#define INV_T    10.0f      // 1/TEMPERATURE
#define DOMW     2.0f
#define EPS_L    1e-8f
#define NORM_EPS 1e-12f

union ABFrag { v16h h; f32x4 f[2]; };

// ---------------------------------------------------------------------------
// Kernel 1: row-normalize feats -> f16 matrix in workspace; zero accumulators.
// One wave (32 lanes) per row of 128 floats (4 per lane).
// ---------------------------------------------------------------------------
__global__ __launch_bounds__(256)
void dacl_normalize_kernel(const float* __restrict__ feats,
                           _Float16* __restrict__ Xh,
                           float* __restrict__ pos_sum,
                           float* __restrict__ neg_sum, int N) {
    const int wave = threadIdx.x >> 5;
    const int lane = threadIdx.x & 31;
    const int row  = blockIdx.x * 8 + wave;

    const float* rp = feats + (size_t)row * DCONST + lane * 4;
    f32x4 v = *(const f32x4*)rp;
    float s = v[0]*v[0] + v[1]*v[1] + v[2]*v[2] + v[3]*v[3];
#pragma unroll
    for (int m = 16; m >= 1; m >>= 1) s += __shfl_xor(s, m, 32);
    const float scale = 1.0f / fmaxf(sqrtf(s), NORM_EPS);

    h16x4 o;
    o[0] = (_Float16)(v[0] * scale);
    o[1] = (_Float16)(v[1] * scale);
    o[2] = (_Float16)(v[2] * scale);
    o[3] = (_Float16)(v[3] * scale);
    *(h16x4*)(Xh + (size_t)row * DCONST + lane * 4) = o;

    const int gid = blockIdx.x * blockDim.x + threadIdx.x;
    if (gid < N) { pos_sum[gid] = 0.0f; neg_sum[gid] = 0.0f; }
}

// ---------------------------------------------------------------------------
// Kernel 2: tiled S = X*X^T via v_wmma_f32_16x16x32_f16, fused exp + masks.
// Block = 256 threads (8 waves). Block tile = 128x128; wave tile = 16x128.
// ---------------------------------------------------------------------------
__global__ __launch_bounds__(256)
void dacl_wmma_kernel(const _Float16* __restrict__ Xh,
                      const int* __restrict__ did,
                      const int* __restrict__ iid,
                      float* __restrict__ pos_sum,
                      float* __restrict__ neg_sum, int N) {
    const int wave  = threadIdx.x >> 5;
    const int lane  = threadIdx.x & 31;
    const int half  = lane >> 4;    // 0: lanes 0-15, 1: lanes 16-31
    const int l15   = lane & 15;
    const int i0    = blockIdx.y * 128 + wave * 16;
    const int j0b   = blockIdx.x * 128;

    // --- A fragments: 16x32 f16, lane = M (l15), K split per half-wave ---
    // lane L: dwords 0-3 hold K = kk + half*8 + [0..8), dwords 4-7 hold +16.
    ABFrag A[4];
    {
        const _Float16* aptr = Xh + (size_t)(i0 + l15) * DCONST + half * 8;
#pragma unroll
        for (int k4 = 0; k4 < 4; ++k4) {
            A[k4].f[0] = *(const f32x4*)(aptr + k4 * 32);
            A[k4].f[1] = *(const f32x4*)(aptr + k4 * 32 + 16);
        }
    }

    // Row metadata for this lane's C rows: M = r + 8*half.
    int didI[8], iidI[8];
#pragma unroll
    for (int r = 0; r < 8; ++r) {
        const int gi = i0 + r + 8 * half;
        didI[r] = did[gi];
        iidI[r] = iid[gi];
    }

    float posAcc[8], negAcc[8];
#pragma unroll
    for (int r = 0; r < 8; ++r) { posAcc[r] = 0.0f; negAcc[r] = 0.0f; }

#pragma unroll
    for (int t = 0; t < 8; ++t) {
        const int j0 = j0b + t * 16;

        // --- B fragments: 32x16 f16, lane = N (l15), K = kk + half*8 + [0..8), then +16 ---
        const _Float16* bptr = Xh + (size_t)(j0 + l15) * DCONST + half * 8;
        ABFrag Bf[4];
#pragma unroll
        for (int k4 = 0; k4 < 4; ++k4) {
            Bf[k4].f[0] = *(const f32x4*)(bptr + k4 * 32);
            Bf[k4].f[1] = *(const f32x4*)(bptr + k4 * 32 + 16);
        }

        v8f c = {0.f, 0.f, 0.f, 0.f, 0.f, 0.f, 0.f, 0.f};
#pragma unroll
        for (int k4 = 0; k4 < 4; ++k4) {
            c = __builtin_amdgcn_wmma_f32_16x16x32_f16(
                    /*neg_a=*/false, A[k4].h, /*neg_b=*/false, Bf[k4].h,
                    /*c_mod=*/(short)0, c, /*reuse_a=*/false, /*reuse_b=*/false);
        }

        const int gj = j0 + l15;
        const int dj = did[gj];
        const int ij = iid[gj];

#pragma unroll
        for (int r = 0; r < 8; ++r) {
            const int gi = i0 + r + 8 * half;
            const float e = __expf(c[r] * INV_T);
            const bool sameD = (didI[r] == dj);
            const bool pos   = sameD && (iidI[r] == ij) && (gi != gj);
            // weight: 2 on cross-dataset, 1 on same-dataset non-positive
            // (diagonal included), 0 on positives.
            const float w = sameD ? (pos ? 0.0f : 1.0f) : DOMW;
            posAcc[r] += pos ? e : 0.0f;
            negAcc[r] += e * w;
        }
    }

    // Reduce over the 16 columns held by each half-wave, one atomic per row.
#pragma unroll
    for (int r = 0; r < 8; ++r) {
        float p = posAcc[r], q = negAcc[r];
#pragma unroll
        for (int m = 8; m >= 1; m >>= 1) {
            p += __shfl_xor(p, m, 32);
            q += __shfl_xor(q, m, 32);
        }
        if (l15 == 0) {
            const int gi = i0 + r + 8 * half;
            atomicAdd(&pos_sum[gi], p);
            atomicAdd(&neg_sum[gi], q);
        }
    }
}

// ---------------------------------------------------------------------------
// Kernel 3: per-row loss, masked mean with cnt==0 fallback. Single block.
// ---------------------------------------------------------------------------
__global__ __launch_bounds__(256)
void dacl_finalize_kernel(const float* __restrict__ pos_sum,
                          const float* __restrict__ neg_sum,
                          float* __restrict__ out, int N) {
    __shared__ float sv[256];
    __shared__ float sc[256];
    __shared__ float sa[256];

    float lv = 0.0f, la = 0.0f, lc = 0.0f;
    for (int r = threadIdx.x; r < N; r += 256) {
        const float ps = pos_sum[r];
        const float ns = neg_sum[r];
        const float loss = -logf(ps / (ps + ns + EPS_L));
        la += loss;
        if (ps > 0.0f) { lv += loss; lc += 1.0f; }
    }
    sv[threadIdx.x] = lv;
    sc[threadIdx.x] = lc;
    sa[threadIdx.x] = la;
    __syncthreads();

    for (int s = 128; s > 0; s >>= 1) {
        if (threadIdx.x < s) {
            sv[threadIdx.x] += sv[threadIdx.x + s];
            sc[threadIdx.x] += sc[threadIdx.x + s];
            sa[threadIdx.x] += sa[threadIdx.x + s];
        }
        __syncthreads();
    }

    if (threadIdx.x == 0) {
        const float cnt = sc[0];
        out[0] = (cnt > 0.0f) ? (sv[0] / fmaxf(cnt, 1.0f))
                              : (sa[0] / (float)N);
    }
}

// ---------------------------------------------------------------------------
extern "C" void kernel_launch(void* const* d_in, const int* in_sizes, int n_in,
                              void* d_out, int out_size, void* d_ws, size_t ws_size,
                              hipStream_t stream) {
    const float* feats = (const float*)d_in[0];
    const int*   did   = (const int*)d_in[1];
    const int*   iid   = (const int*)d_in[2];
    float*       out   = (float*)d_out;

    const int N = in_sizes[1];            // B*L = 8192
    // Workspace layout: [N*128 f16 Xh][N f32 pos_sum][N f32 neg_sum]
    _Float16* Xh      = (_Float16*)d_ws;
    float*    pos_sum = (float*)((char*)d_ws + (size_t)N * DCONST * sizeof(_Float16));
    float*    neg_sum = pos_sum + N;

    // Kernel 1: normalize + stage f16 + zero accumulators.
    dacl_normalize_kernel<<<dim3(N / 8), dim3(256), 0, stream>>>(
        feats, Xh, pos_sum, neg_sum, N);

    // Kernel 2: WMMA similarity + fused exp/masks/row-sums.
    dacl_wmma_kernel<<<dim3(N / 128, N / 128), dim3(256), 0, stream>>>(
        Xh, did, iid, pos_sum, neg_sum, N);

    // Kernel 3: scalar loss.
    dacl_finalize_kernel<<<dim3(1), dim3(256), 0, stream>>>(
        pos_sum, neg_sum, out, N);
}
